// Model_13408887898332
// MI455X (gfx1250) — matrix operand order repaired and checked by reading, not compiled, not run on hardware
//
#include <hip/hip_runtime.h>
#include <hip/hip_fp16.h>

// ---------------------------------------------------------------------------
// MI455X / gfx1250 implementation of the deformable-conv CNN reference.
//   - pack_conv_w16 / pack_fc_w16: convert f32 weights to zero-padded f16
//     layouts so WMMA A-fragments are two contiguous 16B loads, no guards.
//   - tap_gemm_kernel<DEFORM>: per-tap channel GEMM via v_wmma_f32_16x16x32_f16
//     (offset conv = integer taps, deform conv = bilinear-sampled taps).
//     One wave = one 16(cout) x 16(pixel) tile. Branch-free fragment builds.
//   - bn_leaky_pool_kernel: BN + LeakyReLU + 2x2 avgpool fused.
//   - fc_wmma_kernel: FC layers via WMMA; activation vector staged into LDS
//     with the Tensor Data Mover (tensor_load_to_lds + s_wait_tensorcnt).
// ---------------------------------------------------------------------------

#define NEG_SLOPE 0.01f
#define BN_EPS    1e-5f

typedef __attribute__((ext_vector_type(16))) _Float16 v16h;
typedef __attribute__((ext_vector_type(8)))  _Float16 v8h;
typedef __attribute__((ext_vector_type(8)))  float    v8f;
typedef __attribute__((ext_vector_type(4)))  float    f32x4;
typedef unsigned int u32x4 __attribute__((ext_vector_type(4)));
typedef int          i32x8 __attribute__((ext_vector_type(8)));
typedef int          i32x4 __attribute__((ext_vector_type(4)));

__device__ __forceinline__ float leaky_f(float x) {
    return x >= 0.f ? x : NEG_SLOPE * x;
}
__device__ __forceinline__ int imin_(int a, int b) { return a < b ? a : b; }
__device__ __forceinline__ int imax_(int a, int b) { return a > b ? a : b; }

// ---------------------------------------------------------------------------
// Weight packers (zero-padded f16).
//   conv: W[Cout,Cin,16] f32  ->  WT[16][Mp][Cp] f16   (Mp=ceil16, Cp=ceil32)
//   fc:   W[dout,din]    f32  ->  WT[Mp][DinPad] f16
// ---------------------------------------------------------------------------
__global__ void pack_conv_w16(const float* __restrict__ W, _Float16* __restrict__ O,
                              int Cout, int Cin, int Mp, int Cp)
{
    const int i = blockIdx.x * blockDim.x + threadIdx.x;
    const int total = 16 * Mp * Cp;
    if (i >= total) return;
    const int k = i / (Mp * Cp);
    const int r = i - k * Mp * Cp;
    const int m = r / Cp;
    const int c = r - m * Cp;
    const float v = (m < Cout && c < Cin) ? W[(m * Cin + c) * 16 + k] : 0.f;
    O[i] = (_Float16)v;
}

__global__ void pack_fc_w16(const float* __restrict__ W, _Float16* __restrict__ O,
                            int dout, int din, int Mp, int DinPad)
{
    const size_t i = (size_t)blockIdx.x * blockDim.x + threadIdx.x;
    const size_t total = (size_t)Mp * DinPad;
    if (i >= total) return;
    const int m = (int)(i / DinPad);
    const int c = (int)(i - (size_t)m * DinPad);
    const float v = (m < dout && c < din) ? W[(size_t)m * din + c] : 0.f;
    O[i] = (_Float16)v;
}

// ---------------------------------------------------------------------------
// Tap-GEMM kernel.  OUT[co,p] = bias[co] + sum_{k,ci} W[co,ci,k] * S_k[ci,p].
//   X   : [Cin, H, W]      f32 input activation
//   OFF : [32, P]          f32 offsets (2k = dy, 2k+1 = dx), null if !DEFORM
//   WT  : [16][Mp][Cp]     f16 packed weights (zero-padded)
//   OUT : [Cout, P]        f32, P = (H-1)*(W-1)   (K=4, PAD=1 -> Ho = H-1)
// grid.x covers pixel tiles (8 waves/block), grid.y covers Mp/16 cout tiles.
// ---------------------------------------------------------------------------
template <bool DEFORM>
__global__ __launch_bounds__(256)
void tap_gemm_kernel(const float* __restrict__ X,
                     const float* __restrict__ OFF,
                     const _Float16* __restrict__ WT,
                     const float* __restrict__ Bs,
                     float* __restrict__ OUT,
                     int Cin, int Cp, int H, int W, int Cout, int Mp)
{
    const int Ho = H - 1, Wo = W - 1;
    const int P  = Ho * Wo;
    const int HW = H * W;
    const int lane = threadIdx.x & 31;
    const int wave = threadIdx.x >> 5;
    const int hi   = lane >> 4;       // K half-select (wave32 WMMA layout)
    const int n    = lane & 15;       // column / row-in-tile index

    const int pixTile = blockIdx.x * 8 + wave;
    const int p0 = pixTile * 16;
    if (p0 >= P) return;              // wave-uniform exit
    const int coutBase = blockIdx.y * 16;

    int p = p0 + n;
    const bool pOK = p < P;
    if (!pOK) p = P - 1;              // clamp for address formation only
    const int ho = p / Wo;
    const int wo = p - ho * Wo;

    const int mRow = coutBase + n;    // < Mp by construction (padded weights)

    v8f acc = {0.f, 0.f, 0.f, 0.f, 0.f, 0.f, 0.f, 0.f};

#pragma unroll 1
    for (int k = 0; k < 16; ++k) {
        const int kh = k >> 2, kw = k & 3;
        float w00 = 0.f, w01 = 0.f, w10 = 0.f, w11 = 0.f;
        int   i00 = 0,   i01 = 0,   i10 = 0,   i11 = 0;

        if (DEFORM) {
            const float dy = OFF[(2 * k)     * P + p];
            const float dx = OFF[(2 * k + 1) * P + p];
            const float py = (float)(ho - 1 + kh) + dy;
            const float px = (float)(wo - 1 + kw) + dx;
            const float y0f = floorf(py), x0f = floorf(px);
            const float ty = py - y0f,    tx = px - x0f;
            const int y0 = (int)y0f, x0 = (int)x0f;
            const int y1 = y0 + 1,   x1 = x0 + 1;
            const float vy0 = (y0 >= 0 && y0 < H) ? 1.f : 0.f;
            const float vy1 = (y1 >= 0 && y1 < H) ? 1.f : 0.f;
            const float vx0 = (x0 >= 0 && x0 < W) ? 1.f : 0.f;
            const float vx1 = (x1 >= 0 && x1 < W) ? 1.f : 0.f;
            w00 = (1.f - ty) * (1.f - tx) * vy0 * vx0;
            w01 = (1.f - ty) * tx         * vy0 * vx1;
            w10 = ty         * (1.f - tx) * vy1 * vx0;
            w11 = ty         * tx         * vy1 * vx1;
            const int y0c = imin_(imax_(y0, 0), H - 1);
            const int y1c = imin_(imax_(y1, 0), H - 1);
            const int x0c = imin_(imax_(x0, 0), W - 1);
            const int x1c = imin_(imax_(x1, 0), W - 1);
            i00 = y0c * W + x0c;  i01 = y0c * W + x1c;
            i10 = y1c * W + x0c;  i11 = y1c * W + x1c;
        } else {
            const int yc = ho - 1 + kh;
            const int xc = wo - 1 + kw;
            const bool valid = (yc >= 0 && yc < H && xc >= 0 && xc < W);
            w00 = valid ? 1.f : 0.f;
            i00 = imin_(imax_(yc, 0), H - 1) * W + imin_(imax_(xc, 0), W - 1);
        }

        const _Float16* wrowk = WT + ((size_t)k * Mp + mRow) * Cp;

#pragma unroll 1
        for (int c0 = 0; c0 < Cp; c0 += 32) {
            // A fragment: element j holds W[mRow][K], K = j+16*hi,
            // i.e. one contiguous run of 16 halves -> two 16B loads, no guards
            // (weights are zero-padded to [Mp][Cp]).
            const v8h a0 = *(const v8h*)(wrowk + c0 + 16 * hi);
            const v8h a1 = *(const v8h*)(wrowk + c0 + 16 * hi + 8);
            v16h a;
#pragma unroll
            for (int j = 0; j < 8; ++j) { a[j] = a0[j]; a[8 + j] = a1[j]; }

            // B fragment: element j holds S[c0+16*hi+j][n].  Loads are
            // unconditional from channel-clamped addresses; padding channels
            // are zeroed by select (branch-free).
            v16h b;
#pragma unroll
            for (int j = 0; j < 16; ++j) {
                const int cB = c0 + 16 * hi + j;
                const float* xc_ = X + (size_t)imin_(cB, Cin - 1) * HW;
                float s;
                if (DEFORM) {
                    s = w00 * xc_[i00] + w01 * xc_[i01]
                      + w10 * xc_[i10] + w11 * xc_[i11];
                } else {
                    s = w00 * xc_[i00];
                }
                b[j] = (_Float16)(cB < Cin ? s : 0.f);
            }
            acc = __builtin_amdgcn_wmma_f32_16x16x32_f16(
                      false, a, false, b, (short)0, acc, false, false);
        }
    }

    // D layout: VGPR r holds row (r + 8*hi), column = lane&15 = pixel p0+n.
#pragma unroll
    for (int r = 0; r < 8; ++r) {
        const int row = coutBase + r + 8 * hi;
        if (row < Cout && pOK)
            OUT[(size_t)row * P + (p0 + n)] = acc[r] + Bs[row];
    }
}

// ---------------------------------------------------------------------------
// BN + LeakyReLU + 2x2 avg pool.  PRE: [C, Ho, Wo] -> OUT: [C, Ho/2, Wo/2]
// ---------------------------------------------------------------------------
__global__ void bn_leaky_pool_kernel(const float* __restrict__ PRE,
                                     float* __restrict__ OUT,
                                     const float* __restrict__ g,
                                     const float* __restrict__ b,
                                     const float* __restrict__ m,
                                     const float* __restrict__ v,
                                     int C, int Ho, int Wo)
{
    const int Hp = Ho >> 1, Wp = Wo >> 1;
    const int total = C * Hp * Wp;
    const int i = blockIdx.x * blockDim.x + threadIdx.x;
    if (i >= total) return;
    const int c  = i / (Hp * Wp);
    const int r  = i - c * Hp * Wp;
    const int hp = r / Wp;
    const int wp = r - hp * Wp;
    const float sc = g[c] * rsqrtf(v[c] + BN_EPS);
    const float sh = b[c] - m[c] * sc;
    const float* base = PRE + (size_t)c * Ho * Wo;
    float sum = 0.f;
#pragma unroll
    for (int oy = 0; oy < 2; ++oy)
#pragma unroll
        for (int ox = 0; ox < 2; ++ox) {
            float t = base[(2 * hp + oy) * Wo + (2 * wp + ox)];
            sum += leaky_f(t * sc + sh);
        }
    OUT[i] = 0.25f * sum;
}

// ---------------------------------------------------------------------------
// FC layer via WMMA: y[dout] = W16[dout,din] @ x + b  (+ optional BN+leaky).
// W16 is zero-padded f16 [Mp][DinPad]; x chunk staged into LDS with the TDM.
// Block = 256 threads = 8 waves; each wave owns 16 output rows.
// ---------------------------------------------------------------------------
#define FC_CHUNK 2048

__global__ __launch_bounds__(256)
void fc_wmma_kernel(const _Float16* __restrict__ W16, const float* __restrict__ Bs,
                    const float* __restrict__ g,  const float* __restrict__ bb,
                    const float* __restrict__ mm, const float* __restrict__ vv,
                    const float* __restrict__ x,  float* __restrict__ y,
                    int din, int DinPad, int dout, int Mp, int doAct)
{
    __shared__ float xs[FC_CHUNK];
    const int lane = threadIdx.x & 31;
    const int wave = threadIdx.x >> 5;
    const int hi   = lane >> 4;
    const int mBase = (blockIdx.x * 8 + wave) * 16;
    const bool waveActive = mBase < Mp;
    const int mRow = imin_(mBase + (lane & 15), Mp - 1);

    v8f acc = {0.f, 0.f, 0.f, 0.f, 0.f, 0.f, 0.f, 0.f};

    for (int c0 = 0; c0 < din; c0 += FC_CHUNK) {
        const int nch = imin_(FC_CHUNK, din - c0);
        const int nchPad = (nch + 31) & ~31;

#if defined(__gfx1250__) && __has_builtin(__builtin_amdgcn_tensor_load_to_lds)
        // --- Tensor Data Mover: DMA x[c0 .. c0+nch) into LDS (one issuer). ---
        if (threadIdx.x == 0) {
            const unsigned long long ga  = (unsigned long long)(x + c0);
            const unsigned           lds = (unsigned)(unsigned long long)&xs[0];
            const unsigned long long nd  = (unsigned long long)(unsigned)nch;
            union { unsigned w[4]; } g0u;
            g0u.w[0] = 1u;                                   // count=1, user D#
            g0u.w[1] = lds;                                  // lds_addr
            g0u.w[2] = (unsigned)(ga & 0xffffffffull);       // global_addr lo
            g0u.w[3] = (unsigned)((ga >> 32) & 0x1ffffffull) // global_addr hi
                     | (2u << 30);                           // type=2 (image)
            union { unsigned long long q[4]; int w[8]; } g1u;
            g1u.q[0] = (2ull << 16)                          // data_size = 4B
                     | ((nd & 0xffffull) << 48);             // tensor_dim0 lo16
            g1u.q[1] = (nd >> 16)                            // tensor_dim0 hi16
                     | (1ull << 16)                          // tensor_dim1 = 1
                     | ((nd & 0xffffull) << 48);             // tile_dim0 = nch
            g1u.q[2] = 1ull                                  // tile_dim1 = 1
                     | (nd << 32);                           // dim0_stride = nch
            g1u.q[3] = 0ull;
            u32x4 g0v; i32x8 g1v;
            for (int t = 0; t < 4; ++t) g0v[t] = g0u.w[t];
            for (int t = 0; t < 8; ++t) g1v[t] = g1u.w[t];
            i32x4 z4 = {0, 0, 0, 0};
            i32x8 z8 = {0, 0, 0, 0, 0, 0, 0, 0};
            // 6-arg form (amdgpu-toolchain clang): (g0, g1, g2, g3, g4, cpol)
            __builtin_amdgcn_tensor_load_to_lds(g0v, g1v, z4, z4, z8, 0);
            __builtin_amdgcn_s_wait_tensorcnt((short)0);
        }
        // Zero-pad the LDS tail so the K loop needs no per-element guards.
        for (int i = nch + threadIdx.x; i < FC_CHUNK; i += 256) xs[i] = 0.f;
        __syncthreads();
#else
        for (int i = threadIdx.x; i < nch; i += 256) xs[i] = x[c0 + i];
        for (int i = nch + threadIdx.x; i < FC_CHUNK; i += 256) xs[i] = 0.f;
        __syncthreads();
#endif

        if (waveActive) {
            const _Float16* wrow = W16 + (size_t)mRow * DinPad + c0;
            __builtin_prefetch(wrow + FC_CHUNK, 0, 1);
#pragma unroll 1
            for (int kk = 0; kk < nchPad; kk += 32) {
                // A: two contiguous 16B f16 loads (zero-padded weights).
                const v8h a0 = *(const v8h*)(wrow + kk + 16 * hi);
                const v8h a1 = *(const v8h*)(wrow + kk + 16 * hi + 8);
                v16h a;
#pragma unroll
                for (int j = 0; j < 8; ++j) { a[j] = a0[j]; a[8 + j] = a1[j]; }

                // B: 16 consecutive floats from LDS as 4x b128, broadcast
                // across all N columns (zero-padded tail in LDS).
                v16h b;
#pragma unroll
                for (int q = 0; q < 4; ++q) {
                    const f32x4 xv = *(const f32x4*)&xs[kk + 16 * hi + 4 * q];
#pragma unroll
                    for (int t = 0; t < 4; ++t)
                        b[4 * q + t] = (_Float16)xv[t];
                }
                acc = __builtin_amdgcn_wmma_f32_16x16x32_f16(
                          false, a, false, b, (short)0, acc, false, false);
            }
        }
        __syncthreads();
    }

    // All N columns of D are identical (B broadcast); lanes with N==0 store.
    if (waveActive && (lane & 15) == 0) {
#pragma unroll
        for (int r = 0; r < 8; ++r) {
            const int row = mBase + r + 8 * hi;
            if (row < dout) {
                float val = acc[r] + Bs[row];
                if (doAct) {
                    val = (val - mm[row]) * g[row] * rsqrtf(vv[row] + BN_EPS)
                        + bb[row];
                    val = leaky_f(val);
                }
                y[row] = val;
            }
        }
    }
}

// ---------------------------------------------------------------------------
// Host-side orchestration.
// d_in layout (setup_inputs insertion order):
//   [0]  x [3*1024*1024]
//   per stage s (base = 1 + 8*s):
//     +0 off_w [32*Cin*16]  +1 off_b [32]  +2 dc_w [Cout*Cin*16]  +3 dc_b [Cout]
//     +4 bn.g  +5 bn.b  +6 bn.m  +7 bn.v          (each [Cout])
//   fc layer i (base = 41 + 6*i for i<3; 59 for i==3):
//     +0 w [dout*din]  +1 b [dout]  (+2..+5 bn g,b,m,v for i<3)
// ---------------------------------------------------------------------------
extern "C" void kernel_launch(void* const* d_in, const int* in_sizes, int n_in,
                              void* d_out, int out_size, void* d_ws, size_t ws_size,
                              hipStream_t stream)
{
    (void)in_sizes; (void)n_in; (void)out_size; (void)ws_size;
    static const int CHs[6]   = {3, 128, 96, 64, 32, 16};
    static const int FDIMS[5] = {15376, 4096, 512, 64, 7};

    char* ws = (char*)d_ws;
    float*    offb   = (float*)   (ws + (size_t)0x00000000); // <= 134 MB
    float*    preb   = (float*)   (ws + (size_t)0x09000000); // <= 536 MB
    float*    actA   = (float*)   (ws + (size_t)0x2B000000); // <= 134 MB
    float*    actB   = (float*)   (ws + (size_t)0x34000000); // <= 134 MB
    float*    fcA    = (float*)   (ws + (size_t)0x3D000000);
    float*    fcB    = (float*)   (ws + (size_t)0x3D004000);
    _Float16* wt16o  = (_Float16*)(ws + (size_t)0x3D010000); // offset-conv w, 2MB
    _Float16* wt16d  = (_Float16*)(ws + (size_t)0x3D210000); // deform-conv w, 2MB
    _Float16* w16fc  = (_Float16*)(ws + (size_t)0x3E000000); // fc weights, 128MB

    const float* cur = (const float*)d_in[0];
    float* pools[2] = {actA, actB};
    int H = 1024;

    for (int s = 0; s < 5; ++s) {
        const int Cin = CHs[s], Cout = CHs[s + 1];
        const int base = 1 + 8 * s;
        const float* off_w = (const float*)d_in[base + 0];
        const float* off_b = (const float*)d_in[base + 1];
        const float* dc_w  = (const float*)d_in[base + 2];
        const float* dc_b  = (const float*)d_in[base + 3];
        const float* bn_g  = (const float*)d_in[base + 4];
        const float* bn_b  = (const float*)d_in[base + 5];
        const float* bn_m  = (const float*)d_in[base + 6];
        const float* bn_v  = (const float*)d_in[base + 7];

        const int Cp   = (Cin + 31) & ~31;
        const int MpO  = 32;                   // offset conv Cout = 32
        const int MpD  = (Cout + 15) & ~15;

        const int nO = 16 * MpO * Cp;
        pack_conv_w16<<<(nO + 255) / 256, 256, 0, stream>>>(off_w, wt16o, 32, Cin, MpO, Cp);
        const int nD = 16 * MpD * Cp;
        pack_conv_w16<<<(nD + 255) / 256, 256, 0, stream>>>(dc_w, wt16d, Cout, Cin, MpD, Cp);

        const int Ho = H - 1, Wo = H - 1;
        const int P  = Ho * Wo;
        const int pixTiles = (P + 15) / 16;
        const dim3 blk(256);

        dim3 gOff((pixTiles + 7) / 8, MpO / 16);
        tap_gemm_kernel<false><<<gOff, blk, 0, stream>>>(
            cur, nullptr, wt16o, off_b, offb, Cin, Cp, H, H, 32, MpO);

        dim3 gDef((pixTiles + 7) / 8, MpD / 16);
        tap_gemm_kernel<true><<<gDef, blk, 0, stream>>>(
            cur, offb, wt16d, dc_b, preb, Cin, Cp, H, H, Cout, MpD);

        const int Hp = Ho / 2;
        const int tot = Cout * Hp * Hp;
        bn_leaky_pool_kernel<<<(tot + 255) / 256, 256, 0, stream>>>(
            preb, pools[s & 1], bn_g, bn_b, bn_m, bn_v, Cout, Ho, Wo);

        cur = pools[s & 1];
        H = Hp;
    }

    // FC head: [16*31*31] = 15376 (layout already matches reshape order).
    const float* fin = cur;
    float* fouts[4] = {fcA, fcB, fcA, (float*)d_out};
    for (int i = 0; i < 4; ++i) {
        const int base = 41 + 6 * i;          // i==3 -> 59 (w, b only)
        const float* w  = (const float*)d_in[base + 0];
        const float* b  = (const float*)d_in[base + 1];
        const float *g = nullptr, *bb = nullptr, *m = nullptr, *v = nullptr;
        const int act = (i < 3) ? 1 : 0;
        if (act) {
            g  = (const float*)d_in[base + 2];
            bb = (const float*)d_in[base + 3];
            m  = (const float*)d_in[base + 4];
            v  = (const float*)d_in[base + 5];
        }
        const int din = FDIMS[i], dout = FDIMS[i + 1];
        const int Mp = (dout + 15) & ~15;
        const int DinPad = (din + 31) & ~31;

        const size_t nW = (size_t)Mp * DinPad;
        pack_fc_w16<<<(unsigned)((nW + 255) / 256), 256, 0, stream>>>(
            w, w16fc, dout, din, Mp, DinPad);

        const int mtiles = Mp / 16;
        fc_wmma_kernel<<<(mtiles + 7) / 8, 256, 0, stream>>>(
            w16fc, b, g, bb, m, v, fin, fouts[i], din, DinPad, dout, Mp, act);
        fin = fouts[i];
    }
}
